// SlidingWindowAttention_652835029570
// MI455X (gfx1250) — matrix operand order repaired and checked by reading, not compiled, not run on hardware
//
#include <hip/hip_runtime.h>
#include <hip/hip_bf16.h>
#include <stdint.h>

// Problem constants (from reference)
#define BATCH   4
#define SEQ     4096
#define DMODEL  1024
#define NHEADS  16
#define DH      64
#define STRIDE_C 128
#define NW      31
#define MTOT    (BATCH*SEQ)          // 16384 tokens

typedef __bf16 bf16_t;
typedef bf16_t v16bf __attribute__((ext_vector_type(16)));
typedef float  v8f   __attribute__((ext_vector_type(8)));

union FragAB { v16bf v; uint4 q[2]; };

__device__ __forceinline__ unsigned short f2bf(float f) {
  unsigned u = __float_as_uint(f);
  u += 0x7FFFu + ((u >> 16) & 1u);   // round-to-nearest-even
  return (unsigned short)(u >> 16);
}
__device__ __forceinline__ unsigned pack2(float a, float b) {
  return (unsigned)f2bf(a) | ((unsigned)f2bf(b) << 16);
}

// CDNA5 async global->LDS copy of 16 bytes per lane (tracked by ASYNCcnt).
// LDS destination address comes from vdst (low 32 bits of the generic LDS pointer);
// memaddr = 64-bit VGPR pair (GV mode).
__device__ __forceinline__ void async_copy16(void* lds_dst, const void* gsrc) {
  unsigned lds_off = (unsigned)(unsigned long long)lds_dst;
  asm volatile("global_load_async_to_lds_b128 %0, %1, off"
               :: "v"(lds_off), "v"(gsrc) : "memory");
}
__device__ __forceinline__ void wait_async0() {
  asm volatile("s_wait_asynccnt 0x0" ::: "memory");
}

// ---------------------------------------------------------------------------
// Kernel 1a: hidden fp32 -> bf16 X, and zero the ctx accumulator (same count)
// ---------------------------------------------------------------------------
__global__ void cvt_x_zero_ctx(const float* __restrict__ x,
                               unsigned short* __restrict__ xb,
                               float* __restrict__ ctx) {
  const int total = MTOT * DMODEL;   // 16M
  for (int i = blockIdx.x * blockDim.x + threadIdx.x; i < total;
       i += gridDim.x * blockDim.x) {
    xb[i]  = f2bf(x[i]);
    ctx[i] = 0.0f;
  }
}

// ---------------------------------------------------------------------------
// Kernel 1b: W fp32 [K=1024][N=1024] -> bf16 WT [N][K]  (B-fragment wants
// fixed n per lane with contiguous k)
// ---------------------------------------------------------------------------
__global__ void cvt_wT(const float* __restrict__ w,
                       unsigned short* __restrict__ wt) {
  int i = blockIdx.x * 256 + threadIdx.x;      // grid covers exactly 1M
  int n = i >> 10, k = i & 1023;
  wt[i] = f2bf(w[k * DMODEL + n]);
}

// ---------------------------------------------------------------------------
// Tiled WMMA GEMM: C[M=16384,N=1024] = A[M,1024] * B(T stored [N,1024])
// BM=128 BN=128 BK=32, 8 waves, each wave 64x32 (4x2 16x16 frags).
// Double-buffered LDS; bf16 operands staged with GLOBAL_LOAD_ASYNC_TO_LDS_B128
// so the copy of K-step k+1 overlaps the 8 WMMAs of K-step k.
// AMODE 0: A is bf16 (async).  AMODE 1: A is fp32 ctx scaled by 1/count.
// SMODE 0: bf16 out [B,H,S,64]   (Q,K)
// SMODE 1: bf16 out [B,H,64,S]   (V transposed; packed 16B stores)
// SMODE 2: fp32 out [M,1024]     (final output)
// ---------------------------------------------------------------------------
template<int AMODE, int SMODE>
__global__ __launch_bounds__(256) void gemm128(const void* __restrict__ Av,
                                               const unsigned short* __restrict__ BT,
                                               void* __restrict__ Cv) {
  __shared__ unsigned short As[2][128][40];   // pad rows to 80B (16B aligned)
  __shared__ unsigned short Bs[2][128][40];

  const int tid  = threadIdx.x;
  const int m0   = blockIdx.y * 128;
  const int n0   = blockIdx.x * 128;
  const int wid  = tid >> 5, lane = tid & 31;
  const int lr   = lane & 15, hi = lane >> 4;
  const int wm   = (wid >> 2) * 64;
  const int wn   = (wid & 3) * 32;

  v8f acc[4][2];
#pragma unroll
  for (int mt = 0; mt < 4; mt++)
#pragma unroll
    for (int nt = 0; nt < 2; nt++) acc[mt][nt] = {};

  // Stage one 128x32 A tile + 128x32 B tile for K-step k0 into buffer bb.
  auto stage = [&](int k0, int bb) {
#pragma unroll
    for (int c = tid; c < 512; c += 256) {
      int row = c >> 2, seg = c & 3;
      async_copy16(&Bs[bb][row][seg * 8],
                   BT + (n0 + row) * 1024 + k0 + seg * 8);
      if (AMODE == 0) {
        const unsigned short* Ab = (const unsigned short*)Av;
        async_copy16(&As[bb][row][seg * 8],
                     Ab + (m0 + row) * 1024 + k0 + seg * 8);
      } else {
        const float* Af = (const float*)Av;
        int mrow = m0 + row;
        int s = mrow & (SEQ - 1);
        int lo  = (s <= 255) ? 0 : ((s - 128) >> 7);  // first window of token
        int hiw = s >> 7; if (hiw > 30) hiw = 30;     // last window of token
        float scale = 1.0f / ((float)(hiw - lo + 1) + 1e-8f);
        const float* ap = Af + mrow * 1024 + k0 + seg * 8;
        float4 f0 = *(const float4*)(ap);
        float4 f1 = *(const float4*)(ap + 4);
        uint4 va;
        va.x = pack2(f0.x * scale, f0.y * scale);
        va.y = pack2(f0.z * scale, f0.w * scale);
        va.z = pack2(f1.x * scale, f1.y * scale);
        va.w = pack2(f1.z * scale, f1.w * scale);
        *(uint4*)&As[bb][row][seg * 8] = va;
      }
    }
  };

  // Prologue: stage tile 0, wait for async copies, publish to the block.
  stage(0, 0);
  wait_async0();
  __syncthreads();

  int bb = 0;
  for (int k0 = 0; k0 < 1024; k0 += 32) {
    const int nk = k0 + 32;
    if (nk < 1024) stage(nk, bb ^ 1);   // overlap copy of next tile

    // ---- fragments + 8 WMMAs on current buffer ----
    FragAB af[4], bfr[2];
#pragma unroll
    for (int mt = 0; mt < 4; mt++) {
      int m = wm + mt * 16 + lr;
      af[mt].q[0] = *(const uint4*)&As[bb][m][8 * hi];
      af[mt].q[1] = *(const uint4*)&As[bb][m][16 + 8 * hi];
    }
#pragma unroll
    for (int nt = 0; nt < 2; nt++) {
      int n = wn + nt * 16 + lr;
      bfr[nt].q[0] = *(const uint4*)&Bs[bb][n][8 * hi];
      bfr[nt].q[1] = *(const uint4*)&Bs[bb][n][16 + 8 * hi];
    }
#pragma unroll
    for (int mt = 0; mt < 4; mt++)
#pragma unroll
      for (int nt = 0; nt < 2; nt++)
        acc[mt][nt] = __builtin_amdgcn_wmma_f32_16x16x32_bf16(
            false, af[mt].v, false, bfr[nt].v, (short)0, acc[mt][nt],
            false, false);

    if (nk < 1024) wait_async0();       // next tile landed in LDS
    __syncthreads();                    // everyone done with bb, bb^1 ready
    bb ^= 1;
  }

  // ---- store ----
#pragma unroll
  for (int mt = 0; mt < 4; mt++) {
#pragma unroll
    for (int nt = 0; nt < 2; nt++) {
      int nglob = n0 + wn + nt * 16 + lr;
      int mbase = m0 + wm + mt * 16 + 8 * hi;   // C frag: row = i + 8*hi
      if (SMODE == 0) {           // bf16 [B,H,S,64]
        unsigned short* out = (unsigned short*)Cv;
        int h = nglob >> 6, d = nglob & 63;
#pragma unroll
        for (int i = 0; i < 8; i++) {
          int m = mbase + i;
          int b = m >> 12, s = m & (SEQ - 1);
          out[(((b * NHEADS + h) * SEQ + s) << 6) + d] = f2bf(acc[mt][nt][i]);
        }
      } else if (SMODE == 1) {    // bf16 [B,H,64,S], 8 consecutive s -> 16B
        unsigned short* out = (unsigned short*)Cv;
        int h = nglob >> 6, d = nglob & 63;
        int b = mbase >> 12, s = mbase & (SEQ - 1);
        uint4 v;
        v.x = pack2(acc[mt][nt][0], acc[mt][nt][1]);
        v.y = pack2(acc[mt][nt][2], acc[mt][nt][3]);
        v.z = pack2(acc[mt][nt][4], acc[mt][nt][5]);
        v.w = pack2(acc[mt][nt][6], acc[mt][nt][7]);
        *(uint4*)(out + (((b * NHEADS + h) * DH + d) << 12) + s) = v;
      } else {                    // fp32 [M,1024]
        float* out = (float*)Cv;
#pragma unroll
        for (int i = 0; i < 8; i++)
          out[(mbase + i) * 1024 + nglob] = acc[mt][nt][i];
      }
    }
  }
}

// ---------------------------------------------------------------------------
// Attention: one block per (h, window, b). 4 waves, each wave does 4 q-tiles
// of 16 rows. Scores (fp32) live in LDS, overwritten in place by bf16 exp(P).
// ctx accumulated into global fp32 via atomics (<=2 adds/elem, commutative
// => deterministic).
// ---------------------------------------------------------------------------
__global__ __launch_bounds__(128) void attn_kernel(const unsigned short* __restrict__ Qb,
                                                   const unsigned short* __restrict__ Kb,
                                                   const unsigned short* __restrict__ Vt,
                                                   float* __restrict__ ctx) {
  __shared__ float sc[4][16][256];   // 64 KB: per-wave 16x256 fp32 scores
  const int h = blockIdx.x, n = blockIdx.y, b = blockIdx.z;
  const int s0  = n * STRIDE_C;
  const int tid = threadIdx.x, wid = tid >> 5, lane = tid & 31;
  const int lr  = lane & 15, hi = lane >> 4;

  const int bh = b * NHEADS + h;
  const unsigned short* Qh = Qb + bh * SEQ * DH;
  const unsigned short* Kh = Kb + bh * SEQ * DH;
  const unsigned short* Vh = Vt + bh * DH * SEQ;
  float*          smem_f = &sc[wid][0][0];
  unsigned short* smem_h = (unsigned short*)smem_f;
  float*          rsw    = smem_f + 2048;   // row sums; reuses dead score space

  for (int j = 0; j < 4; j++) {
    const int qt   = wid * 4 + j;
    const int qrow = s0 + qt * 16 + lr;

    // Q fragments (16 x 64 = two 16x32 A-frags)
    FragAB aq[2];
#pragma unroll
    for (int f = 0; f < 2; f++) {
      const unsigned short* qp = Qh + qrow * DH + f * 32;
      aq[f].q[0] = *(const uint4*)(qp + 8 * hi);
      aq[f].q[1] = *(const uint4*)(qp + 16 + 8 * hi);
    }

    // ---- pass 1: scores = Q K^T / 8 ----
    for (int kt = 0; kt < 16; kt++) {
      v8f acc = {};
      const int krow = s0 + kt * 16 + lr;       // B-frag: n = key row
#pragma unroll
      for (int f = 0; f < 2; f++) {
        FragAB bk;
        const unsigned short* kp = Kh + krow * DH + f * 32 + 8 * hi;
        bk.q[0] = *(const uint4*)(kp);
        bk.q[1] = *(const uint4*)(kp + 16);
        acc = __builtin_amdgcn_wmma_f32_16x16x32_bf16(
            false, aq[f].v, false, bk.v, (short)0, acc, false, false);
      }
#pragma unroll
      for (int i = 0; i < 8; i++)
        smem_f[(i + 8 * hi) * 256 + kt * 16 + lr] = acc[i] * 0.125f;
    }
    __syncthreads();

    // ---- softmax: lane handles row lr, cols [hi*128, hi*128+128) ----
    const float4* rowp = (const float4*)&smem_f[lr * 256 + hi * 128];
    float lmax = -1e30f;
#pragma unroll 4
    for (int t = 0; t < 32; t++) {
      float4 v = rowp[t];
      lmax = fmaxf(lmax, fmaxf(fmaxf(v.x, v.y), fmaxf(v.z, v.w)));
    }
    lmax = fmaxf(lmax, __shfl_xor(lmax, 16));
    float lsum = 0.f;
    unsigned* pout = (unsigned*)&smem_h[lr * 256 + hi * 128];
    for (int t = 0; t < 32; t++) {
      float4 v = rowp[t];
      float e0 = __expf(v.x - lmax), e1 = __expf(v.y - lmax);
      float e2 = __expf(v.z - lmax), e3 = __expf(v.w - lmax);
      lsum += (e0 + e1) + (e2 + e3);
      pout[t * 2]     = pack2(e0, e1);          // unnormalized P, bf16
      pout[t * 2 + 1] = pack2(e2, e3);
      asm volatile("" ::: "memory");            // keep load->store order
    }
    lsum += __shfl_xor(lsum, 16);
    if (hi == 0) rsw[lr] = lsum;
    __syncthreads();

    float inv[8];
#pragma unroll
    for (int i = 0; i < 8; i++) inv[i] = 1.0f / rsw[i + 8 * hi];

    // ---- pass 2: ctx = (P / rowsum) @ V ----
#pragma unroll
    for (int nt = 0; nt < 4; nt++) {
      v8f acc = {};
      const int d = nt * 16 + lr;
      const unsigned short* vp = Vh + d * SEQ + s0;
#pragma unroll
      for (int kb = 0; kb < 8; kb++) {
        FragAB ap, bv;
        const unsigned short* pp = smem_h + lr * 256 + kb * 32;
        ap.q[0] = *(const uint4*)(pp + 8 * hi);
        ap.q[1] = *(const uint4*)(pp + 16 + 8 * hi);
        const unsigned short* vv = vp + kb * 32 + 8 * hi;
        bv.q[0] = *(const uint4*)(vv);
        bv.q[1] = *(const uint4*)(vv + 16);
        acc = __builtin_amdgcn_wmma_f32_16x16x32_bf16(
            false, ap.v, false, bv.v, (short)0, acc, false, false);
      }
      const int dglob = h * DH + d;
#pragma unroll
      for (int i = 0; i < 8; i++) {
        int s = s0 + qt * 16 + i + 8 * hi;
        atomicAdd(&ctx[(b * SEQ + s) * DMODEL + dglob], acc[i] * inv[i]);
      }
    }
    __syncthreads();
  }
}

// ---------------------------------------------------------------------------
extern "C" void kernel_launch(void* const* d_in, const int* in_sizes, int n_in,
                              void* d_out, int out_size, void* d_ws, size_t ws_size,
                              hipStream_t stream) {
  const float* hid = (const float*)d_in[0];
  const float* Wq  = (const float*)d_in[1];
  const float* Wk  = (const float*)d_in[2];
  const float* Wv  = (const float*)d_in[3];
  const float* Wo  = (const float*)d_in[4];

  // workspace layout (~200 MB)
  char* ws = (char*)d_ws;
  size_t off = 0;
  unsigned short* Xb  = (unsigned short*)(ws + off); off += (size_t)MTOT * DMODEL * 2;
  unsigned short* WqT = (unsigned short*)(ws + off); off += (size_t)DMODEL * DMODEL * 2;
  unsigned short* WkT = (unsigned short*)(ws + off); off += (size_t)DMODEL * DMODEL * 2;
  unsigned short* WvT = (unsigned short*)(ws + off); off += (size_t)DMODEL * DMODEL * 2;
  unsigned short* WoT = (unsigned short*)(ws + off); off += (size_t)DMODEL * DMODEL * 2;
  unsigned short* Qbf = (unsigned short*)(ws + off); off += (size_t)MTOT * DMODEL * 2;
  unsigned short* Kbf = (unsigned short*)(ws + off); off += (size_t)MTOT * DMODEL * 2;
  unsigned short* Vtb = (unsigned short*)(ws + off); off += (size_t)MTOT * DMODEL * 2;
  float*          ctx = (float*)(ws + off);          off += (size_t)MTOT * DMODEL * 4;

  cvt_x_zero_ctx<<<8192, 256, 0, stream>>>(hid, Xb, ctx);
  cvt_wT<<<4096, 256, 0, stream>>>(Wq, WqT);
  cvt_wT<<<4096, 256, 0, stream>>>(Wk, WkT);
  cvt_wT<<<4096, 256, 0, stream>>>(Wv, WvT);
  cvt_wT<<<4096, 256, 0, stream>>>(Wo, WoT);

  dim3 gg(1024 / 128, MTOT / 128);   // (8, 128)
  gemm128<0, 0><<<gg, 256, 0, stream>>>(Xb, WqT, Qbf);
  gemm128<0, 0><<<gg, 256, 0, stream>>>(Xb, WkT, Kbf);
  gemm128<0, 1><<<gg, 256, 0, stream>>>(Xb, WvT, Vtb);

  attn_kernel<<<dim3(NHEADS, NW, BATCH), 128, 0, stream>>>(Qbf, Kbf, Vtb, ctx);

  gemm128<1, 2><<<gg, 256, 0, stream>>>(ctx, WoT, d_out);
}
